// GCNs_4389456576945
// MI455X (gfx1250) — matrix operand order repaired and checked by reading, not compiled, not run on hardware
//
#include <hip/hip_runtime.h>
#include <hip/hip_bf16.h>
#include <cstdint>
#include <cstddef>

// ---------------------------------------------------------------------------
// GCN2 stack on MI455X (gfx1250, wave32). Memory-bound: 9 dense 12288^2
// matvecs, adj (576 MB fp32) >> L2 (192 MB) so it must stream each layer.
// Plan: one fused pass converts adj to bf16 in d_ws AND computes layer 0 in
// fp32 (~864 MB traffic); layers 1..8 are v_wmma_f32_16x16x32_bf16 matvecs
// (8 x 288 MB). Total ~3.17 GB ~= 136 us at 23.3 TB/s vs 225 us all-fp32.
// ---------------------------------------------------------------------------

typedef __bf16 bf16_t;
typedef __bf16 v16bf __attribute__((ext_vector_type(16)));
typedef __bf16 v8bf  __attribute__((ext_vector_type(8)));
typedef __bf16 v4bf  __attribute__((ext_vector_type(4)));
typedef float  v8f   __attribute__((ext_vector_type(8)));

#define GCN_ALPHA 0.3f

__device__ __forceinline__ float act_apply(float t, int act) {
  switch (act) {
    case 1:  return t > 0.0f ? t : 0.01f * t;      // leaky_relu (default slope)
    case 2:  return fmaxf(t, 0.0f);                // relu
    case 3:  return 1.0f / (1.0f + expf(-t));      // sigmoid
    default: return t;                             // identity
  }
}

// ---- Fused: adj fp32->bf16 conversion + layer 0 (h = A@x0, fp32 acc) ------
// Grid n/16 blocks x 256 threads; each wave converts 2 rows (coalesced float4
// reads, b64 bf16 stores) while accumulating the row dot against x0.
__global__ __launch_bounds__(256) void cvt_adj_fused_l0(
    const float* __restrict__ adj, const float* __restrict__ x0,
    const float* __restrict__ W, int n,
    bf16_t* __restrict__ adjb, bf16_t* __restrict__ xout) {
  const int wave = threadIdx.x >> 5;
  const int lane = threadIdx.x & 31;
  #pragma unroll
  for (int rr = 0; rr < 2; ++rr) {
    const int row = blockIdx.x * 16 + wave * 2 + rr;
    const float* src = adj + (long long)row * n;
    bf16_t* dst = adjb + (long long)row * n;
    float acc = 0.0f;
    for (int k = lane * 4; k < n; k += 128) {
      float4 a = *(const float4*)(src + k);
      float4 x = *(const float4*)(x0 + k);
      v4bf o;
      o[0] = (bf16_t)a.x; o[1] = (bf16_t)a.y;
      o[2] = (bf16_t)a.z; o[3] = (bf16_t)a.w;
      *(v4bf*)(dst + k) = o;
      acc += a.x * x.x + a.y * x.y + a.z * x.z + a.w * x.w;
    }
    #pragma unroll
    for (int off = 16; off; off >>= 1) acc += __shfl_xor(acc, off, 32);
    if (lane == 0) {
      const float h = (1.0f - GCN_ALPHA) * acc + GCN_ALPHA * x0[row];
      xout[row] = (bf16_t)(h * W[0]);  // layer 0: no activation
    }
  }
}

__global__ void copy_vec_f32(const float* __restrict__ src,
                             float* __restrict__ dst, int n) {
  int i = blockIdx.x * blockDim.x + threadIdx.x;
  if (i < n) dst[i] = src[i];
}

// ---- WMMA bf16 layer: one block = 16 rows; 8 waves split K; LDS reduce -----
// A layout (16-bit A 16x32): lane l holds row (l&15), K=(l>>4)*16+[0,16)
// -> two b128 loads/lane/step; lanes l and l+16
// jointly consume each 128B cacheline of a row exactly once.
// B (32x16): lane l holds col (l&15), K=(l>>4)*16+[0,16). x is staged in LDS
// once per block and replicated into all 16 B columns (broadcast ds_load,
// conflict-free), so every D column equals y. A-loads are software-pipelined
// one step ahead of the WMMA.
__global__ __launch_bounds__(256) void gcn2_layer_wmma(
    const bf16_t* __restrict__ adjb,
    const bf16_t* __restrict__ xin,
    const float*  __restrict__ x0,
    const float*  __restrict__ W, int layer,
    int n, int act, int is_final,
    bf16_t* __restrict__ xout, int* __restrict__ iout) {
  extern __shared__ bf16_t xs[];          // n bf16 (24 KB for n=12288)
  __shared__ float sm[8][16];

  const int tid     = threadIdx.x;
  const int wave    = tid >> 5;
  const int lane    = tid & 31;
  const int lane_lo = lane & 15;
  const int lane_hi = lane >> 4;
  const int rowbase = blockIdx.x * 16;

  // stage x into LDS: 256 threads x 8 bf16 per iter
  for (int i = tid * 8; i < n; i += 256 * 8)
    *(v8bf*)(xs + i) = *(const v8bf*)(xin + i);
  __syncthreads();

  const int kPerWave = n >> 3;            // 8 waves split K
  const int kBeg  = wave * kPerWave;
  const int steps = kPerWave >> 5;        // 32-K per WMMA

  const bf16_t* arow = adjb + (long long)(rowbase + lane_lo) * n;
  const int aoff = lane_hi * 16;
  const int boff = lane_hi * 16;

  v8f c = {};
  int k = kBeg;
  v8bf a0 = *(const v8bf*)(arow + k + aoff);
  v8bf a1 = *(const v8bf*)(arow + k + aoff + 8);
  for (int s = 1; s < steps; ++s) {
    const int kn = k + 32;
    v8bf na0 = *(const v8bf*)(arow + kn + aoff);      // prefetch next step
    v8bf na1 = *(const v8bf*)(arow + kn + aoff + 8);
    v16bf b = *(const v16bf*)(xs + k + boff);         // LDS broadcast
    v16bf a = __builtin_shufflevector(a0, a1,
        0, 1, 2, 3, 4, 5, 6, 7, 8, 9, 10, 11, 12, 13, 14, 15);
    c = __builtin_amdgcn_wmma_f32_16x16x32_bf16(
        false, a, false, b, (short)0, c, false, false);
    a0 = na0; a1 = na1; k = kn;
  }
  {
    v16bf b = *(const v16bf*)(xs + k + boff);
    v16bf a = __builtin_shufflevector(a0, a1,
        0, 1, 2, 3, 4, 5, 6, 7, 8, 9, 10, 11, 12, 13, 14, 15);
    c = __builtin_amdgcn_wmma_f32_16x16x32_bf16(
        false, a, false, b, (short)0, c, false, false);
  }

  // All D columns identical; lane 0 holds y[rb+v], lane 16 holds y[rb+8+v].
  if (lane_lo == 0) {
    #pragma unroll
    for (int v = 0; v < 8; ++v) sm[wave][lane_hi * 8 + v] = c[v];
  }
  __syncthreads();

  if (tid < 16) {
    float dot = 0.0f;
    #pragma unroll
    for (int w = 0; w < 8; ++w) dot += sm[w][tid];
    const int r = rowbase + tid;
    const float h = (1.0f - GCN_ALPHA) * dot + GCN_ALPHA * x0[r];
    const float t = h * W[layer];
    if (is_final) {
      iout[r] = (int)((float)n / (1.0f + expf(-t)));  // n*sigmoid -> i32 trunc
    } else {
      xout[r] = (bf16_t)act_apply(t, act);
    }
  }
}

// ---- fp32 fallback (ws too small for bf16 adj): 1 wave per row ------------
__global__ __launch_bounds__(256) void gcn2_layer_f32(
    const float* __restrict__ adj, const float* __restrict__ xin,
    const float* __restrict__ x0, const float* __restrict__ W, int layer,
    int n, int act, int is_final,
    float* __restrict__ xout, int* __restrict__ iout) {
  const int wave = threadIdx.x >> 5;
  const int lane = threadIdx.x & 31;
  const int row  = blockIdx.x * 8 + wave;
  if (row >= n) return;
  const float* arow = adj + (long long)row * n;
  float acc = 0.0f;
  for (int k = lane * 4; k < n; k += 128) {
    float4 a = *(const float4*)(arow + k);
    float4 x = *(const float4*)(xin + k);
    acc += a.x * x.x + a.y * x.y + a.z * x.z + a.w * x.w;
  }
  #pragma unroll
  for (int off = 16; off; off >>= 1) acc += __shfl_xor(acc, off, 32);
  if (lane == 0) {
    const float h = (1.0f - GCN_ALPHA) * acc + GCN_ALPHA * x0[row];
    const float t = h * W[layer];
    if (is_final) iout[row] = (int)((float)n / (1.0f + expf(-t)));
    else          xout[row] = act_apply(t, act);
  }
}

extern "C" void kernel_launch(void* const* d_in, const int* in_sizes, int n_in,
                              void* d_out, int out_size, void* d_ws, size_t ws_size,
                              hipStream_t stream) {
  (void)n_in; (void)out_size;
  const float* x0  = (const float*)d_in[0];   // [N,1] fp32
  const float* adj = (const float*)d_in[1];   // [N,N] fp32
  const float* W   = (const float*)d_in[2];   // [9,1,1] fp32
  const int n = in_sizes[0];
  int* iout = (int*)d_out;

  // activation applied to layer L's output (L=8 handled by is_final path)
  static const int acts[9] = {0, 1, 2, 2, 3, 2, 2, 2, 0};

  const long long nn = (long long)n * n;
  const size_t adjBytes = ((size_t)nn * 2 + 255) & ~(size_t)255;
  const size_t vecBytes = ((size_t)n * 2 + 255) & ~(size_t)255;
  const bool bf16_ok = (n % 256 == 0) && ((size_t)n * 2 <= 160 * 1024) &&
                       (ws_size >= adjBytes + 2 * vecBytes);

  if (bf16_ok) {
    bf16_t* adjb = (bf16_t*)d_ws;
    bf16_t* xA = (bf16_t*)((char*)d_ws + adjBytes);
    bf16_t* xB = (bf16_t*)((char*)d_ws + adjBytes + vecBytes);

    // one pass: convert adj to bf16 + compute layer 0 in fp32
    cvt_adj_fused_l0<<<n / 16, 256, 0, stream>>>(adj, x0, W, n, adjb, xA);

    bf16_t* xi = xA;
    bf16_t* xo = xB;
    for (int L = 1; L < 9; ++L) {
      const int fin = (L == 8);
      gcn2_layer_wmma<<<n / 16, 256, (size_t)n * sizeof(bf16_t), stream>>>(
          adjb, xi, x0, W, L, n, acts[L], fin, xo, iout);
      bf16_t* t = xi; xi = xo; xo = t;
    }
  } else {
    // fp32 fallback: only 2*N floats of scratch needed
    float* xA = (float*)d_ws;
    float* xB = xA + n;
    copy_vec_f32<<<(n + 255) / 256, 256, 0, stream>>>(x0, xA, n);
    float* xi = xA;
    float* xo = xB;
    for (int L = 0; L < 9; ++L) {
      const int fin = (L == 8);
      gcn2_layer_f32<<<(n + 7) / 8, 256, 0, stream>>>(
          adj, xi, x0, W, L, n, acts[L], fin, xo, iout);
      float* t = xi; xi = xo; xo = t;
    }
  }
}
